// SparseNNV0_11373073399838
// MI455X (gfx1250) — compile-verified, matrix operand order checked by reading, not run on hardware
//
#include <hip/hip_runtime.h>
#include <hip/hip_bf16.h>

#define BB      4096
#define LL      50
#define TT      4
#define VOCABN  100000
#define HID     505
#define KPADP   512     /* HID padded to multiple of 32 */
#define EDIM    64
#define DENSE_F 13
#define NE      5       /* T+1 embedding slots */
#define OVER_IN 330     /* 64*5 + 10 */
#define FEAT_LD 352     /* OVER_IN padded to multiple of 32 */

typedef __attribute__((ext_vector_type(16))) _Float16 v16h;
typedef __attribute__((ext_vector_type(8)))  _Float16 v8h;
typedef __attribute__((ext_vector_type(8)))  float    v8f;

// ---------------------------------------------------------------------------
// Fragment load from a padded row-major f16 matrix (ld = padded K width).
// ISA 7.12.2 16-bit A 16x32 layout: lane holds M=lane&15; its 16 halves are
// K = kb..kb+7 and kb+16..kb+23 where kb = k0 + 8*(lane>=16).  Both groups
// are contiguous -> two 16-byte vector loads, no predication, no shuffles.
// B fragments use the identical map with the weight matrix pre-transposed to
// [N][K], lane holding N.
// ---------------------------------------------------------------------------
__device__ __forceinline__ v16h frag_ld(const _Float16* __restrict__ base,
                                        int ld, int row, int kb) {
  const _Float16* p = base + (size_t)row * ld + kb;
  v8h lo = *(const v8h*)(p);
  v8h hi = *(const v8h*)(p + 16);
  v16h f;
#pragma unroll
  for (int i = 0; i < 8; ++i) { f[i] = lo[i]; f[i + 8] = hi[i]; }
  return f;
}

// C/D f32 16x16: lane holds N=lane&15; VGPR r holds M = r + 8*(lane>=16).
__device__ __forceinline__ void frag_store(float* __restrict__ C, int ldc,
                                           int row0, int n0, int lane, v8f c,
                                           const float* __restrict__ bias) {
  const int n  = n0 + (lane & 15);
  const int mb = row0 + ((lane >> 4) << 3);
  const float bv = bias[n];
#pragma unroll
  for (int r = 0; r < 8; ++r)
    C[(size_t)(mb + r) * ldc + n] = c[r] + bv;
}

// ---------------------------------------------------------------------------
// Kernel 0: transpose + pad weights to f16 once per call.
//   Wt[t][n][k] (64x512 per table)  <- proj_w[t][k][n], zero for k>=505
//   Ot[n][k]    (64x352)            <- over_w[k][n],    zero for k>=330
// ---------------------------------------------------------------------------
__global__ void __launch_bounds__(256)
k_prep(const float* __restrict__ pw, const float* __restrict__ ow,
       _Float16* __restrict__ Wt, _Float16* __restrict__ Ot) {
  const int NWT = TT * EDIM * KPADP;       // 131072
  const int NOT = EDIM * FEAT_LD;          // 22528
  int gid = blockIdx.x * blockDim.x + threadIdx.x;
  if (gid < NWT) {
    int k = gid & (KPADP - 1);
    int n = (gid >> 9) & (EDIM - 1);
    int t = gid >> 15;
    float v = (k < HID) ? pw[((size_t)t * HID + k) * EDIM + n] : 0.0f;
    Wt[gid] = (_Float16)v;
  } else if (gid < NWT + NOT) {
    int r = gid - NWT;
    int k = r % FEAT_LD;
    int n = r / FEAT_LD;
    float v = (k < OVER_IN) ? ow[(size_t)k * EDIM + n] : 0.0f;
    Ot[r] = (_Float16)v;
  }
}

// ---------------------------------------------------------------------------
// Kernel 1: dense bottom arch -> E[:,0,:]   (4096x13 @ 13x64 + b)
// ---------------------------------------------------------------------------
__global__ void __launch_bounds__(256)
k_dense(const float* __restrict__ dense, const float* __restrict__ dw,
        const float* __restrict__ db, float* __restrict__ E) {
  int gid = blockIdx.x * blockDim.x + threadIdx.x;
  if (gid >= BB * EDIM) return;
  int b = gid >> 6, e = gid & 63;
  float acc = db[e];
#pragma unroll
  for (int f = 0; f < DENSE_F; ++f)
    acc += dense[b * DENSE_F + f] * dw[f * EDIM + e];
  E[(size_t)b * (NE * EDIM) + e] = acc;   // slot 0
}

// ---------------------------------------------------------------------------
// Kernel 2 (dominant, HBM-gather-bound): gather + norm-clip + pool.
// One wave32 per (t,b) bag: lanes stride the 505-float row (coalesced 128B),
// wave shfl_xor reduction for sum-of-squares, prefetch of next gathered row.
// Output written as f16 with K padded to 512 (WMMA-ready A operand).
// ---------------------------------------------------------------------------
__global__ void __launch_bounds__(256)
k_pool(const int* __restrict__ ids, const float* __restrict__ tables,
       _Float16* __restrict__ pooled) {
  const int wave = (blockIdx.x * blockDim.x + threadIdx.x) >> 5;
  const int lane = threadIdx.x & 31;
  if (wave >= TT * BB) return;
  const int t = wave / BB;
  const int b = wave % BB;
  const int*   idp = ids + ((size_t)t * BB + b) * LL;
  const float* tab = tables + (size_t)t * VOCABN * HID;

  float acc[16];
#pragma unroll
  for (int j = 0; j < 16; ++j) acc[j] = 0.0f;

  int idx = idp[0];
  for (int l = 0; l < LL; ++l) {
    const float* row = tab + (size_t)idx * HID;
    if (l + 1 < LL) {                       // prefetch next gathered row
      int nidx = idp[l + 1];
      __builtin_prefetch(tab + (size_t)nidx * HID + lane * 16, 0, 1);
      idx = nidx;
    }
    float v[16];
    float ss = 0.0f;
#pragma unroll
    for (int j = 0; j < 16; ++j) {
      int d = lane + 32 * j;
      v[j] = (d < HID) ? row[d] : 0.0f;     // pad lanes stay exactly 0
      ss += v[j] * v[j];
    }
#pragma unroll
    for (int off = 16; off > 0; off >>= 1)  // wave32 butterfly reduce
      ss += __shfl_xor(ss, off, 32);
    float nrm = sqrtf(ss);
    float sc  = (nrm > 1.0f) ? (1.0f / nrm) : 1.0f;
#pragma unroll
    for (int j = 0; j < 16; ++j) acc[j] += v[j] * sc;
  }

  _Float16* pp = pooled + ((size_t)t * BB + b) * KPADP;
#pragma unroll
  for (int j = 0; j < 16; ++j)
    pp[lane + 32 * j] = (_Float16)acc[j];   // covers all 512 incl. zero pad
}

// ---------------------------------------------------------------------------
// Kernel 3: per-table projection via WMMA: pooled(4096x512) @ Wt^T -> E.
// 4 waves/block, one 16x64 output tile per wave; 16 K-steps of 32.
// ---------------------------------------------------------------------------
__global__ void __launch_bounds__(128)
k_proj(const _Float16* __restrict__ pooled, const _Float16* __restrict__ Wt,
       const float* __restrict__ pb, float* __restrict__ E) {
  const int lane = threadIdx.x & 31;
  const int tile = blockIdx.x * 4 + (threadIdx.x >> 5);
  const int t     = tile >> 8;             // 256 row tiles per table
  const int btile = tile & 255;
  const int row0  = btile * 16;
  const _Float16* A = pooled + (size_t)t * BB * KPADP;
  const _Float16* W = Wt + (size_t)t * EDIM * KPADP;

  const int am = row0 + (lane & 15);
  const int bn = lane & 15;
  const int koff = (lane >> 4) << 3;

  v8f c[4] = {};
  for (int k0 = 0; k0 < KPADP; k0 += 32) {
    v16h a = frag_ld(A, KPADP, am, k0 + koff);
#pragma unroll
    for (int nt = 0; nt < 4; ++nt) {
      v16h bf = frag_ld(W, KPADP, nt * 16 + bn, k0 + koff);
      c[nt] = __builtin_amdgcn_wmma_f32_16x16x32_f16(
          false, a, false, bf, (short)0, c[nt], false, false);
    }
  }
#pragma unroll
  for (int nt = 0; nt < 4; ++nt)
    frag_store(E + (size_t)(1 + t) * EDIM, NE * EDIM, row0, nt * 16, lane,
               c[nt], pb + (size_t)t * EDIM);
}

// ---------------------------------------------------------------------------
// Kernel 4: feats[b] = [E flat (320) | triu(E E^T) (10) | zeros (22)] as f16.
// One wave per sample; 10 wave-reduced dot products of length 64.
// ---------------------------------------------------------------------------
__global__ void __launch_bounds__(32)
k_feats(const float* __restrict__ E, _Float16* __restrict__ feats) {
  const int b    = blockIdx.x;
  const int lane = threadIdx.x;
  const float* e = E + (size_t)b * NE * EDIM;
  _Float16* f = feats + (size_t)b * FEAT_LD;

  float ev[NE][2];
#pragma unroll
  for (int j = 0; j < NE; ++j) {
    ev[j][0] = e[j * EDIM + lane];
    ev[j][1] = e[j * EDIM + lane + 32];
    f[j * EDIM + lane]      = (_Float16)ev[j][0];
    f[j * EDIM + lane + 32] = (_Float16)ev[j][1];
  }
  int p = 0;
#pragma unroll
  for (int i = 0; i < NE; ++i) {
#pragma unroll
    for (int j = i + 1; j < NE; ++j) {      // row-major triu order = jnp.triu_indices
      float s = ev[i][0] * ev[j][0] + ev[i][1] * ev[j][1];
#pragma unroll
      for (int off = 16; off > 0; off >>= 1)
        s += __shfl_xor(s, off, 32);
      if (lane == 0) f[NE * EDIM + p] = (_Float16)s;
      ++p;
    }
  }
  if (lane < FEAT_LD - OVER_IN) f[OVER_IN + lane] = (_Float16)0.0f;  // zero K-pad
}

// ---------------------------------------------------------------------------
// Kernel 5: over arch via WMMA: feats(4096x352) @ Ot^T + b -> out (4096x64).
// ---------------------------------------------------------------------------
__global__ void __launch_bounds__(128)
k_over(const _Float16* __restrict__ feats, const _Float16* __restrict__ Ot,
       const float* __restrict__ ob, float* __restrict__ out) {
  const int lane = threadIdx.x & 31;
  const int tile = blockIdx.x * 4 + (threadIdx.x >> 5);
  const int row0 = tile * 16;

  const int am = row0 + (lane & 15);
  const int bn = lane & 15;
  const int koff = (lane >> 4) << 3;

  v8f c[4] = {};
  for (int k0 = 0; k0 < FEAT_LD; k0 += 32) {
    v16h a = frag_ld(feats, FEAT_LD, am, k0 + koff);
#pragma unroll
    for (int nt = 0; nt < 4; ++nt) {
      v16h bf = frag_ld(Ot, FEAT_LD, nt * 16 + bn, k0 + koff);
      c[nt] = __builtin_amdgcn_wmma_f32_16x16x32_f16(
          false, a, false, bf, (short)0, c[nt], false, false);
    }
  }
#pragma unroll
  for (int nt = 0; nt < 4; ++nt)
    frag_store(out, EDIM, row0, nt * 16, lane, c[nt], ob);
}

// ---------------------------------------------------------------------------
extern "C" void kernel_launch(void* const* d_in, const int* in_sizes, int n_in,
                              void* d_out, int out_size, void* d_ws, size_t ws_size,
                              hipStream_t stream) {
  const float* dense = (const float*)d_in[0];
  const int*   ids   = (const int*)  d_in[1];
  const float* tabs  = (const float*)d_in[2];
  const float* pw    = (const float*)d_in[3];
  const float* pb    = (const float*)d_in[4];
  const float* dw    = (const float*)d_in[5];
  const float* db    = (const float*)d_in[6];
  const float* ow    = (const float*)d_in[7];
  const float* ob    = (const float*)d_in[8];
  float* out = (float*)d_out;

  // Workspace carve (256B aligned): pooled_h | E | feats_h | Wt | Ot
  char* ws = (char*)d_ws;
  const size_t pooled_b = (((size_t)TT * BB * KPADP * 2) + 255) & ~(size_t)255;
  const size_t e_b      = (((size_t)BB * NE * EDIM * 4) + 255) & ~(size_t)255;
  const size_t feats_b  = (((size_t)BB * FEAT_LD * 2) + 255) & ~(size_t)255;
  const size_t wt_b     = (((size_t)TT * EDIM * KPADP * 2) + 255) & ~(size_t)255;
  _Float16* pooled = (_Float16*)ws;
  float*    E      = (float*)   (ws + pooled_b);
  _Float16* feats  = (_Float16*)(ws + pooled_b + e_b);
  _Float16* Wt     = (_Float16*)(ws + pooled_b + e_b + feats_b);
  _Float16* Ot     = (_Float16*)(ws + pooled_b + e_b + feats_b + wt_b);
  (void)in_sizes; (void)n_in; (void)out_size; (void)ws_size;

  const int NPREP = TT * EDIM * KPADP + EDIM * FEAT_LD;
  k_prep <<<(NPREP + 255) / 256,        256, 0, stream>>>(pw, ow, Wt, Ot);
  k_dense<<<(BB * EDIM + 255) / 256,    256, 0, stream>>>(dense, dw, db, E);
  k_pool <<<(TT * BB) / 8,              256, 0, stream>>>(ids, tabs, pooled);
  k_proj <<<TT * (BB / 16) / 4,         128, 0, stream>>>(pooled, Wt, pb, E);
  k_feats<<<BB,                          32, 0, stream>>>(E, feats);
  k_over <<<(BB / 16) / 4,              128, 0, stream>>>(feats, Ot, ob, out);
}
